// MoELayer_70334384439369
// MI455X (gfx1250) — compile-verified, matrix operand order checked by reading, not run on hardware
//
#include <hip/hip_runtime.h>
#include <cstdint>
#include <cstddef>

// Problem sizes (fixed by the reference)
constexpr int kH = 2048;   // hidden
constexpr int kF = 4096;   // intermediate
constexpr int kE = 8;      // experts
constexpr int kT = 8192;   // tokens = 4*2048
constexpr float kAux = 0.01f;
constexpr float kZ   = 0.001f;
constexpr int KC = 256;    // K-chunk staged to LDS by the TDM

typedef __bf16 bf16_t;
typedef __attribute__((ext_vector_type(16))) __bf16 v16bf;
typedef __attribute__((ext_vector_type(8)))  float  v8f;
typedef uint32_t u32x4 __attribute__((ext_vector_type(4)));
typedef uint32_t u32x8 __attribute__((ext_vector_type(8)));

// -------- workspace layout (bytes) --------
constexpr size_t WS_ACC  = 0;                 // 32 floats of loss accumulators
constexpr size_t WS_TOPK = 1024;              // int[T*2] top-2 expert ids
constexpr size_t WS_RW   = 131072;            // float[T*2] routing weights
constexpr size_t WS_HBUF = (size_t)1 << 20;   // bf16[T*F] gate*up intermediate (64 MB, fits L2)

// ---------------- zero init ----------------
__global__ __launch_bounds__(256) void k_zero(float* __restrict__ out, float* __restrict__ acc) {
  size_t i = (size_t)blockIdx.x * blockDim.x + threadIdx.x;
  size_t stride = (size_t)gridDim.x * blockDim.x;
  for (size_t j = i; j < (size_t)kT * kH; j += stride) out[j] = 0.f;
  if (i < 32) acc[i] = 0.f;
}

// ---------------- router ----------------
__global__ __launch_bounds__(256) void k_router(const float* __restrict__ x,
                                                const float* __restrict__ wr,
                                                int*   __restrict__ topk,
                                                float* __restrict__ rwb,
                                                float* __restrict__ acc) {
  int t = blockIdx.x * blockDim.x + threadIdx.x;
  if (t >= kT) return;
  float l[kE];
#pragma unroll
  for (int e = 0; e < kE; ++e) l[e] = 0.f;
  const float* xp = x + (size_t)t * kH;
  for (int h = 0; h < kH; h += 4) {
    float4 xv = *(const float4*)(xp + h);
#pragma unroll
    for (int e = 0; e < kE; ++e) {
      float4 wv = *(const float4*)(wr + (size_t)e * kH + h);
      l[e] += xv.x * wv.x + xv.y * wv.y + xv.z * wv.z + xv.w * wv.w;
    }
  }
  float m = l[0];
#pragma unroll
  for (int e = 1; e < kE; ++e) m = fmaxf(m, l[e]);
  float p[kE]; float s = 0.f;
#pragma unroll
  for (int e = 0; e < kE; ++e) { p[e] = __expf(l[e] - m); s += p[e]; }
  float inv = 1.f / s;
#pragma unroll
  for (int e = 0; e < kE; ++e) p[e] *= inv;
  float z = m + __logf(s);
  atomicAdd(&acc[0], z * z);
#pragma unroll
  for (int e = 0; e < kE; ++e) atomicAdd(&acc[1 + e], p[e]);
  // top-2 (ties -> lowest index, matching lax.top_k)
  int i0 = 0; float p0 = p[0];
#pragma unroll
  for (int e = 1; e < kE; ++e) if (p[e] > p0) { p0 = p[e]; i0 = e; }
  int i1 = -1; float p1 = -1.f;
#pragma unroll
  for (int e = 0; e < kE; ++e) if (e != i0 && p[e] > p1) { p1 = p[e]; i1 = e; }
  float denom = p0 + p1;
  topk[t * 2 + 0] = i0;       topk[t * 2 + 1] = i1;
  rwb [t * 2 + 0] = p0/denom; rwb [t * 2 + 1] = p1/denom;
  atomicAdd(&acc[9 + i0], 1.f);
  atomicAdd(&acc[9 + i1], 1.f);
}

// ---------------- loss finalize ----------------
__global__ void k_loss(const float* __restrict__ acc, float* __restrict__ out) {
  if (threadIdx.x == 0 && blockIdx.x == 0) {
    float aux = 0.f;
#pragma unroll
    for (int e = 0; e < kE; ++e)
      aux += (acc[9 + e] / (float)kT) * (acc[1 + e] / (float)kT);
    aux *= kAux * (float)kE;
    float zl = kZ * acc[0] / (float)kT;
    out[(size_t)kT * kH] = aux + zl;
  }
}

// ---------------- fragment load: 16 contiguous f32 -> v16bf ----------------
__device__ __forceinline__ v16bf load16_f32_bf16(const float* p) {
  v16bf r;
#pragma unroll
  for (int i = 0; i < 16; i += 4) {
    float4 f = *(const float4*)(p + i);
    r[i + 0] = (bf16_t)f.x; r[i + 1] = (bf16_t)f.y;
    r[i + 2] = (bf16_t)f.z; r[i + 3] = (bf16_t)f.w;
  }
  return r;
}

// ---------------- TDM: DMA a [tile_d1 x tile_d0] fp32 tile from global to LDS ----------------
// D# per CDNA5 ISA. 2D tensor -> groups 2/3 disabled.
__device__ __forceinline__ void tdm_load_2d(const float* gptr, uint32_t lds_byte_off,
                                            uint32_t dim0, uint32_t dim1,
                                            uint32_t tile_d0, uint32_t tile_d1) {
  uint64_t ga = (uint64_t)(uintptr_t)gptr;
  u32x4 g0;
  g0.x = 1u;                                                 // count=1, user descriptor
  g0.y = lds_byte_off;                                       // lds_addr
  g0.z = (uint32_t)ga;                                       // global_addr[31:0]
  g0.w = (uint32_t)((ga >> 32) & 0x01FFFFFFu) | (2u << 30);  // global_addr[56:32] | type=2
  u32x8 g1;
  g1[0] = (2u << 16);                                        // data_size=2 (4 bytes)
  g1[1] = (dim0 & 0xFFFFu) << 16;                            // tensor_dim0[15:0]
  g1[2] = (dim0 >> 16) | ((dim1 & 0xFFFFu) << 16);           // tensor_dim0[31:16] | tensor_dim1[15:0]
  g1[3] = ((dim1 >> 16) & 0xFFFFu) | (tile_d0 << 16);        // tensor_dim1[31:16] | tile_dim0
  g1[4] = tile_d1 & 0xFFFFu;                                 // tile_dim1 (tile_dim2=0)
  g1[5] = dim0;                                              // tensor_dim0_stride[31:0]
  g1[6] = 0u;                                                // stride0[47:32] | stride1[15:0]
  g1[7] = 0u;
  asm volatile("tensor_load_to_lds %0, %1" :: "s"(g0), "s"(g1) : "memory");
}

// ---------------- gate/up GEMMs + silu*up, per expert ----------------
// Block: 512 tokens x 16 F-cols. TDM double-buffers 16x256 fp32 stripes of
// Wg/Wu into LDS (TENSORcnt); waves read B fragments via ds_load and stream
// A from global. Each wave: 64 tokens x 16 cols (4 M-subtiles).
__global__ __launch_bounds__(256) void k_gateup(const float* __restrict__ x,
                                                const float* __restrict__ wg,
                                                const float* __restrict__ wu,
                                                bf16_t* __restrict__ hbuf) {
  __shared__ float sw[2][2][16][KC];   // [buf][gate|up][row][col] = 64 KB

  const int lane = threadIdx.x & 31;
  const int wave = threadIdx.x >> 5;
  const int half = lane >> 4;
  const int lidx = lane & 15;
  const int tilesN = kF / 16;
  const int tileN  = (blockIdx.x % tilesN) * 16;
  const int tileM  = (blockIdx.x / tilesN) * 512 + wave * 64;

  const float* ap = x + (size_t)(tileM + lidx) * kH + half * 16;
  const size_t mstep = (size_t)16 * kH;
  const float* wgs = wg + (size_t)tileN * kH;   // 16-row weight stripes for this block
  const float* wus = wu + (size_t)tileN * kH;

  constexpr int nch = kH / KC;   // 8 chunks
  if (wave == 0) {               // preload chunk 0 into buffer 0
    tdm_load_2d(wgs, (uint32_t)(uintptr_t)&sw[0][0][0][0], kH, kF, KC, 16);
    tdm_load_2d(wus, (uint32_t)(uintptr_t)&sw[0][1][0][0], kH, kF, KC, 16);
  }

  v8f cg[4] = {}; v8f cu[4] = {};
  for (int c = 0; c < nch; ++c) {
    if (wave == 0) __builtin_amdgcn_s_wait_tensorcnt(0);   // chunk c landed in LDS
    __syncthreads();
    if (wave == 0 && c + 1 < nch) {                        // overlap DMA of c+1 with compute of c
      const int nb = (c + 1) & 1;
      tdm_load_2d(wgs + (c + 1) * KC, (uint32_t)(uintptr_t)&sw[nb][0][0][0], kH, kF, KC, 16);
      tdm_load_2d(wus + (c + 1) * KC, (uint32_t)(uintptr_t)&sw[nb][1][0][0], kH, kF, KC, 16);
    }
    const int cb = c & 1;
    const int k0 = c * KC;
    for (int kk = 0; kk < KC; kk += 32) {
      __builtin_prefetch(ap + k0 + kk + 128, 0, 1);        // A stream prefetch
      v16bf fg = load16_f32_bf16(&sw[cb][0][lidx][kk + half * 16]);  // ds_load_b128 x4
      v16bf fu = load16_f32_bf16(&sw[cb][1][lidx][kk + half * 16]);
#pragma unroll
      for (int i = 0; i < 4; ++i) {
        v16bf a = load16_f32_bf16(ap + k0 + kk + i * mstep);
        cg[i] = __builtin_amdgcn_wmma_f32_16x16x32_bf16(false, a, false, fg, (short)0, cg[i], false, false);
        cu[i] = __builtin_amdgcn_wmma_f32_16x16x32_bf16(false, a, false, fu, (short)0, cu[i], false, false);
      }
    }
    __syncthreads();   // done reading buf (c&1) before TDM reuses it at chunk c+2
  }
#pragma unroll
  for (int i = 0; i < 4; ++i) {
#pragma unroll
    for (int v = 0; v < 8; ++v) {
      int row = tileM + i * 16 + v + 8 * half;  // C layout: M = v (+8 upper lane half)
      int col = tileN + lidx;                   //           N = lane % 16
      float g = cg[i][v], u = cu[i][v];
      float hv = g * __builtin_amdgcn_rcpf(1.f + __expf(-g)) * u;   // silu(g)*u
      hbuf[(size_t)row * kF + col] = (bf16_t)hv;
    }
  }
}

// ---------------- down GEMM + combine-weighted accumulate, per expert ----------------
// Block: 512 tokens x 16 H-cols. TDM double-buffers 16x256 fp32 stripes of Wd.
__global__ __launch_bounds__(256) void k_down(const bf16_t* __restrict__ hbuf,
                                              const float* __restrict__ wd,
                                              const int*   __restrict__ topk,
                                              const float* __restrict__ rwb,
                                              float* __restrict__ out, int e) {
  __shared__ float sw[2][16][KC];   // 32 KB

  const int lane = threadIdx.x & 31;
  const int wave = threadIdx.x >> 5;
  const int half = lane >> 4;
  const int lidx = lane & 15;
  const int tilesN = kH / 16;
  const int tileN  = (blockIdx.x % tilesN) * 16;
  const int tileM  = (blockIdx.x / tilesN) * 512 + wave * 64;

  const bf16_t* ap = hbuf + (size_t)(tileM + lidx) * kF + half * 16;
  const size_t mstep = (size_t)16 * kF;
  const float* wds = wd + (size_t)tileN * kF;

  constexpr int nch = kF / KC;   // 16 chunks
  if (wave == 0)
    tdm_load_2d(wds, (uint32_t)(uintptr_t)&sw[0][0][0], kF, kH, KC, 16);

  v8f c4[4] = {};
  for (int c = 0; c < nch; ++c) {
    if (wave == 0) __builtin_amdgcn_s_wait_tensorcnt(0);
    __syncthreads();
    if (wave == 0 && c + 1 < nch)
      tdm_load_2d(wds + (c + 1) * KC, (uint32_t)(uintptr_t)&sw[(c + 1) & 1][0][0], kF, kH, KC, 16);
    const int cb = c & 1;
    const int k0 = c * KC;
    for (int kk = 0; kk < KC; kk += 32) {
      __builtin_prefetch(ap + k0 + kk + 256, 0, 1);   // bf16 A stream prefetch
      v16bf b = load16_f32_bf16(&sw[cb][lidx][kk + half * 16]);
#pragma unroll
      for (int i = 0; i < 4; ++i) {
        v16bf a = *(const v16bf*)(ap + k0 + kk + i * mstep);   // h bf16, 32B aligned
        c4[i] = __builtin_amdgcn_wmma_f32_16x16x32_bf16(false, a, false, b, (short)0, c4[i], false, false);
      }
    }
    __syncthreads();
  }
#pragma unroll
  for (int i = 0; i < 4; ++i) {
#pragma unroll
    for (int v = 0; v < 8; ++v) {
      int row = tileM + i * 16 + v + 8 * half;
      int col = tileN + lidx;
      int2   ti = ((const int2*)  topk)[row];
      float2 wv = ((const float2*)rwb )[row];
      float scale = (ti.x == e) ? wv.x : ((ti.y == e) ? wv.y : 0.f);
      // experts are launched sequentially on the stream -> plain RMW is race-free
      out[(size_t)row * kH + col] += scale * c4[i][v];
    }
  }
}

// ---------------- launch ----------------
extern "C" void kernel_launch(void* const* d_in, const int* in_sizes, int n_in,
                              void* d_out, int out_size, void* d_ws, size_t ws_size,
                              hipStream_t stream) {
  const float* x  = (const float*)d_in[0];   // [4,2048,2048]
  const float* wr = (const float*)d_in[1];   // [8,2048]
  const float* wg = (const float*)d_in[2];   // [8,4096,2048]
  const float* wu = (const float*)d_in[3];   // [8,4096,2048]
  const float* wd = (const float*)d_in[4];   // [8,2048,4096]
  float* out = (float*)d_out;                // [T*H] output + [1] router_loss

  char* ws = (char*)d_ws;
  float*  acc  = (float*) (ws + WS_ACC);
  int*    topk = (int*)   (ws + WS_TOPK);
  float*  rwb  = (float*) (ws + WS_RW);
  bf16_t* hbuf = (bf16_t*)(ws + WS_HBUF);

  k_zero  <<<1024, 256, 0, stream>>>(out, acc);
  k_router<<<kT / 256, 256, 0, stream>>>(x, wr, topk, rwb, acc);
  k_loss  <<<1, 1, 0, stream>>>(acc, out);

  const int gu_blocks = (kT / 512) * (kF / 16);  // 16 * 256 = 4096 blocks
  const int dn_blocks = (kT / 512) * (kH / 16);  // 16 * 128 = 2048 blocks
  for (int e = 0; e < kE; ++e) {
    k_gateup<<<gu_blocks, 256, 0, stream>>>(x,
                                            wg + (size_t)e * kF * kH,
                                            wu + (size_t)e * kF * kH,
                                            hbuf);
    k_down  <<<dn_blocks, 256, 0, stream>>>(hbuf,
                                            wd + (size_t)e * kH * kF,
                                            topk, rwb, out, e);
  }
}
